// MultiScaleSpectralBlock_22428319219748
// MI455X (gfx1250) — matrix operand order repaired and checked by reading, not compiled, not run on hardware
//
#include <hip/hip_runtime.h>
#include <math.h>

// ---------------------------------------------------------------------------
// MI455X (gfx1250, wave32) implementation of MultiScaleSpectralBlock forward.
// Dense channel-mixing convs (1x1 / 5x5 / 3x3) run as implicit GEMMs on
// v_wmma_f32_16x16x32_bf16 (fp32 accumulate), 4 accumulators per wave
// (16 co x 64 pix) to amortize weight-fragment loads. GEMM inputs live in a
// pixel-major bf16 mirror XT[q][c] so each lane's B fragment is a contiguous
// 32B load and A fragments are contiguous 16B loads. Concats materialize for
// free as channel strips of wide XT buffers. im2col tap decode is carried
// incrementally (no integer division in the k-loop); KSZ/pad mode are
// template constants. Depthwise/dilated convs, groupnorms, gates etc. are
// bandwidth-bound VALU kernels with fused epilogues.
// ---------------------------------------------------------------------------

#define HWPIX 16384   // 128*128
#define QTOT  65536   // 4*128*128
#define NB    4

typedef __bf16 bf16;
typedef __attribute__((ext_vector_type(16))) __bf16 v16bf;
typedef __attribute__((ext_vector_type(8)))  float  v8f;

__device__ __forceinline__ float geluf(float x) {
  return 0.5f * x * (1.0f + erff(x * 0.70710678118654752f));
}
__device__ __forceinline__ int refl128(int v) {
  v = v < 0 ? -v : v;
  return v > 127 ? 254 - v : v;
}

union frag16 { uint4 u[2]; v16bf v; };

// Shared epilogue: bias + residual, fp32 NCHW store + bf16 pixel-major store.
__device__ __forceinline__ void gemm_epilogue(
    v8f acc, int q, int co0, int lane,
    const float* __restrict__ bias, const float* __restrict__ resid,
    float* __restrict__ outF, int outFCtot,
    bf16* __restrict__ outT, int ldt, int toff)
{
  const int n  = q >> 14;
  const int hw = q & 16383;
  const int mbase = (lane >> 4) << 3;      // rows 0..7 or 8..15
  float vals[8];
#pragma unroll
  for (int r = 0; r < 8; ++r) {
    const int c = co0 + mbase + r;
    float v = acc[r];
    if (bias)  v += bias[c];
    if (resid) v += resid[((size_t)n * 64 + c) * HWPIX + hw];
    vals[r] = v;
  }
  if (outF) {
#pragma unroll
    for (int r = 0; r < 8; ++r)
      outF[((size_t)n * outFCtot + co0 + mbase + r) * HWPIX + hw] = vals[r];
  }
  if (outT) {
    union { uint4 u; bf16 hx[8]; } pk;
#pragma unroll
    for (int r = 0; r < 8; ++r) pk.hx[r] = (bf16)vals[r];
    *(uint4*)(outT + (size_t)q * ldt + (toff + co0 + mbase)) = pk.u;
  }
}

// ---------------------------------------------------------------------------
// 1x1-conv GEMM:  D[co, q] = sum_k A[co, k] * XT[q, boff+k]
// Wave tile: 16 co x 64 pixels (4 WMMA accumulators). Block = 4 waves.
// ---------------------------------------------------------------------------
__global__ __launch_bounds__(128) void gemm1x1_wmma_kernel(
    const bf16* __restrict__ A, const bf16* __restrict__ Bt,
    int ldb, int boff, int K,
    const float* __restrict__ bias, const float* __restrict__ resid,
    float* __restrict__ outF, int outFCtot,
    bf16* __restrict__ outT, int ldt, int toff)
{
  const int lane = threadIdx.x & 31;
  const int wv   = threadIdx.x >> 5;
  const int q0   = blockIdx.x * 256 + wv * 64;
  const int co0  = blockIdx.y * 16;

  const int m    = lane & 15;
  const int kb   = (lane >> 4) << 4;       // A half: 0 or 16
  const int col  = lane & 15;
  const int ksel = (lane >> 4) << 4;       // B half: 0 or 16

  v8f acc[4];
#pragma unroll
  for (int j = 0; j < 4; ++j) acc[j] = (v8f){0.f,0.f,0.f,0.f,0.f,0.f,0.f,0.f};

  const bf16* arow = A + (size_t)(co0 + m) * K + kb;
  const bf16* brow = Bt + (size_t)(q0 + col) * ldb + boff + ksel;
  const size_t bstep = (size_t)16 * ldb;   // 16 pixels

  for (int k0 = 0; k0 < K; k0 += 32) {
    frag16 ua;
    ua.u[0] = *(const uint4*)(arow + k0);
    ua.u[1] = *(const uint4*)(arow + k0 + 8);
#pragma unroll
    for (int j = 0; j < 4; ++j) {
      frag16 ub;
      const bf16* bp = brow + (size_t)j * bstep + k0;
      ub.u[0] = *(const uint4*)(bp);
      ub.u[1] = *(const uint4*)(bp + 8);
      acc[j] = __builtin_amdgcn_wmma_f32_16x16x32_bf16(
          false, ua.v, false, ub.v, (short)0, acc[j], false, false);
    }
  }
#pragma unroll
  for (int j = 0; j < 4; ++j)
    gemm_epilogue(acc[j], q0 + j * 16 + col, co0, lane,
                  bias, resid, outF, outFCtot, outT, ldt, toff);
}

// ---------------------------------------------------------------------------
// im2col GEMM for dense KSZxKSZ convs. K = Cin*KSZ*KSZ, tap-major.
// Tap/ci decode is carried incrementally -> no division in the k-loop.
// PADMODE: 1 = reflect, 2 = zero.
// ---------------------------------------------------------------------------
template <int KSZ, int PADMODE>
__global__ __launch_bounds__(128) void gemm_im2col_wmma_kernel(
    const bf16* __restrict__ A, const bf16* __restrict__ Bt,
    int ldb, int boff, int K, int Cin, int dil, int pad,
    const float* __restrict__ bias, const float* __restrict__ resid,
    float* __restrict__ outF, int outFCtot,
    bf16* __restrict__ outT, int ldt, int toff)
{
  const int lane = threadIdx.x & 31;
  const int wv   = threadIdx.x >> 5;
  const int q0   = blockIdx.x * 256 + wv * 64;
  const int co0  = blockIdx.y * 16;

  const int m    = lane & 15;
  const int kb   = (lane >> 4) << 4;
  const int col  = lane & 15;
  const int ksel = (lane >> 4) << 4;

  v8f acc[4];
#pragma unroll
  for (int j = 0; j < 4; ++j) acc[j] = (v8f){0.f,0.f,0.f,0.f,0.f,0.f,0.f,0.f};

  const bf16* arow = A + (size_t)(co0 + m) * K + kb;

  // wave tile lives entirely inside one image (HW=16384, tiles 256-aligned)
  const int n = q0 >> 14;
  const bf16* bimg = Bt + ((size_t)n << 14) * ldb + boff;
  int hj[4], wj[4];
#pragma unroll
  for (int j = 0; j < 4; ++j) {
    const int hw = (q0 + j * 16 + col) & 16383;
    hj[j] = hw >> 7; wj[j] = hw & 127;
  }

  // incremental (ci, tx, ty) decode of kst = k0 + ksel
  int ci = ksel, tx = 0, ty = 0;

  for (int k0 = 0; k0 < K; k0 += 32) {
    frag16 ua;
    ua.u[0] = *(const uint4*)(arow + k0);
    ua.u[1] = *(const uint4*)(arow + k0 + 8);

    const int dy = ty * dil - pad;
    const int dx = tx * dil - pad;
#pragma unroll
    for (int j = 0; j < 4; ++j) {
      int hh = hj[j] + dy;
      int ww = wj[j] + dx;
      bool oob = false;
      if (PADMODE == 1) { hh = refl128(hh); ww = refl128(ww); }
      else {
        oob = (hh < 0) | (hh > 127) | (ww < 0) | (ww > 127);
        hh = hh < 0 ? 0 : (hh > 127 ? 127 : hh);
        ww = ww < 0 ? 0 : (ww > 127 ? 127 : ww);
      }
      const bf16* bp = bimg + (size_t)((hh << 7) + ww) * ldb + ci;
      const uint4 z = {0u, 0u, 0u, 0u};
      frag16 ub;
      ub.u[0] = oob ? z : *(const uint4*)(bp);
      ub.u[1] = oob ? z : *(const uint4*)(bp + 8);
      acc[j] = __builtin_amdgcn_wmma_f32_16x16x32_bf16(
          false, ua.v, false, ub.v, (short)0, acc[j], false, false);
    }
    ci += 32;
    if (ci >= Cin) {            // at most one tap step per iteration (Cin>=32)
      ci -= Cin;
      if (++tx == KSZ) { tx = 0; ++ty; }
    }
  }
#pragma unroll
  for (int j = 0; j < 4; ++j)
    gemm_epilogue(acc[j], q0 + j * 16 + col, co0, lane,
                  bias, resid, outF, outFCtot, outT, ldt, toff);
}

// ---------------------------------------------------------------------------
// GroupNorm: one block per (n, group); two-pass; optional act + dual outputs.
// act: 0=none, 1=gelu, 2=leaky(0.01)
// ---------------------------------------------------------------------------
__global__ __launch_bounds__(256) void groupnorm_kernel(
    const float* __restrict__ in, int C, int cpg,
    const float* __restrict__ gamma, const float* __restrict__ beta, int act,
    float* __restrict__ outF, bf16* __restrict__ outT, int ldt, int toff)
{
  const int groups = C / cpg;
  const int g = blockIdx.x % groups;
  const int n = blockIdx.x / groups;
  const size_t base = ((size_t)n * C + (size_t)g * cpg) * HWPIX;
  const int cnt = cpg * HWPIX;

  float s = 0.f, ss = 0.f;
  for (int i = threadIdx.x; i < cnt; i += 256) {
    const float v = in[base + i]; s += v; ss += v * v;
  }
  __shared__ float rs[256], rq[256];
  rs[threadIdx.x] = s; rq[threadIdx.x] = ss; __syncthreads();
  for (int o = 128; o > 0; o >>= 1) {
    if (threadIdx.x < o) { rs[threadIdx.x] += rs[threadIdx.x + o];
                           rq[threadIdx.x] += rq[threadIdx.x + o]; }
    __syncthreads();
  }
  const float mu  = rs[0] / (float)cnt;
  const float var = rq[0] / (float)cnt - mu * mu;
  const float inv = rsqrtf(var + 1e-5f);

  for (int i = threadIdx.x; i < cnt; i += 256) {
    const int c  = g * cpg + (i >> 14);
    const int hw = i & 16383;
    float v = (in[base + i] - mu) * inv * gamma[c] + beta[c];
    if (act == 1)      v = geluf(v);
    else if (act == 2) v = v > 0.f ? v : 0.01f * v;
    if (outF) outF[base + i] = v;
    if (outT) outT[((size_t)n * HWPIX + hw) * ldt + toff + c] = (bf16)v;
  }
}

// ---------------------------------------------------------------------------
// Depthwise (dilated) conv, fused residual/scale, dual outputs.
// ---------------------------------------------------------------------------
__global__ __launch_bounds__(256) void dwconv_kernel(
    const float* __restrict__ in, int C,
    const float* __restrict__ w, const float* __restrict__ bias,
    int ksz, int dil, int pad, int reflect,
    const float* __restrict__ resid, const float* __restrict__ scaleNC,
    float* __restrict__ outF, bf16* __restrict__ outT, int ldt, int toff)
{
  const int idx = blockIdx.x * 256 + threadIdx.x;
  if (idx >= NB * C * HWPIX) return;
  const int hw = idx & 16383;
  const int c  = (idx >> 14) % C;
  const int n  = idx / (C * HWPIX);
  const int h = hw >> 7, wp = hw & 127;
  const float* wc  = w + c * ksz * ksz;
  const float* inc = in + ((size_t)n * C + c) * HWPIX;
  float acc = bias ? bias[c] : 0.f;
  for (int ky = 0; ky < ksz; ++ky) {
    int hh = h + ky * dil - pad;
    if (reflect) hh = refl128(hh);
    for (int kx = 0; kx < ksz; ++kx) {
      int wx = wp + kx * dil - pad;
      if (reflect) wx = refl128(wx);
      float v = 0.f;
      if (reflect || ((unsigned)hh < 128u && (unsigned)wx < 128u))
        v = inc[(hh << 7) + wx];
      acc += wc[ky * ksz + kx] * v;
    }
  }
  if (resid)   acc += resid[((size_t)n * C + c) * HWPIX + hw];
  if (scaleNC) acc *= scaleNC[n * C + c];
  if (outF) outF[((size_t)n * C + c) * HWPIX + hw] = acc;
  if (outT) outT[((size_t)n * HWPIX + hw) * ldt + toff + c] = (bf16)acc;
}

// ---------------------------------------------------------------------------
// Small VALU 1x1 conv (16-ch bottleneck chain, pa0). act: 0 none,1 gelu,2 prelu
// ---------------------------------------------------------------------------
__global__ __launch_bounds__(256) void conv1x1_small_kernel(
    const float* __restrict__ in, int Cin, int Cout,
    const float* __restrict__ w, const float* __restrict__ bias,
    int act, const float* __restrict__ prelu_a,
    float* __restrict__ outF, bf16* __restrict__ outT, int ldt, int toff)
{
  const int idx = blockIdx.x * 256 + threadIdx.x;
  if (idx >= NB * Cout * HWPIX) return;
  const int hw = idx & 16383;
  const int co = (idx >> 14) % Cout;
  const int n  = idx / (Cout * HWPIX);
  const float* wrow = w + co * Cin;
  const float* inn  = in + (size_t)n * Cin * HWPIX + hw;
  float acc = bias ? bias[co] : 0.f;
  for (int ci = 0; ci < Cin; ++ci) acc += wrow[ci] * inn[(size_t)ci * HWPIX];
  if (act == 1) acc = geluf(acc);
  else if (act == 2) { const float a = prelu_a[0]; acc = acc > 0.f ? acc : a * acc; }
  if (outF) outF[((size_t)n * Cout + co) * HWPIX + hw] = acc;
  if (outT) outT[((size_t)n * HWPIX + hw) * ldt + toff + co] = (bf16)acc;
}

// Global average pool: one block per (n,c).
__global__ __launch_bounds__(256) void gap_kernel(
    const float* __restrict__ in, float* __restrict__ out)
{
  const size_t base = (size_t)blockIdx.x * HWPIX;
  float s = 0.f;
  for (int i = threadIdx.x; i < HWPIX; i += 256) s += in[base + i];
  __shared__ float r[256];
  r[threadIdx.x] = s; __syncthreads();
  for (int o = 128; o > 0; o >>= 1) {
    if (threadIdx.x < o) r[threadIdx.x] += r[threadIdx.x + o];
    __syncthreads();
  }
  if (threadIdx.x == 0) out[blockIdx.x] = r[0] * (1.f / (float)HWPIX);
}

// Dynamic-filter generator: 1x1 conv (64->576) on gap, GN(4), softmax over taps
__global__ void freq_filter_kernel(
    const float* __restrict__ gapv, const float* __restrict__ fw_w,
    const float* __restrict__ gn_g, const float* __restrict__ gn_b,
    float* __restrict__ fw_out)   // [N][64][9]
{
  const int n = blockIdx.x;
  const int c = threadIdx.x;      // 0..575
  __shared__ float sv[576];
  __shared__ float smu[4], sinv[4];
  const float* g = gapv + n * 64;
  float acc = 0.f;
  for (int ci = 0; ci < 64; ++ci) acc += fw_w[c * 64 + ci] * g[ci];
  sv[c] = acc; __syncthreads();
  if (c < 4) {
    float s = 0.f, ss = 0.f;
    for (int i = 0; i < 144; ++i) { const float v = sv[c * 144 + i]; s += v; ss += v * v; }
    const float mu = s / 144.f;
    smu[c] = mu; sinv[c] = rsqrtf(ss / 144.f - mu * mu + 1e-5f);
  }
  __syncthreads();
  const int grp = c / 144;
  const float vn = (sv[c] - smu[grp]) * sinv[grp] * gn_g[c] + gn_b[c];
  __syncthreads();
  sv[c] = vn; __syncthreads();
  if (c < 64) {
    float mx = -1e30f;
    for (int t = 0; t < 9; ++t) mx = fmaxf(mx, sv[c * 9 + t]);
    float e[9], se = 0.f;
    for (int t = 0; t < 9; ++t) { e[t] = expf(sv[c * 9 + t] - mx); se += e[t]; }
    const float r = 1.f / se;
    for (int t = 0; t < 9; ++t) fw_out[((size_t)n * 64 + c) * 9 + t] = e[t] * r;
  }
}

// low = 9-tap dynamic filter (reflect pad 1); high = x - low.
__global__ __launch_bounds__(256) void lowhigh_kernel(
    const float* __restrict__ x, const float* __restrict__ fw,
    float* __restrict__ low, float* __restrict__ high)
{
  const int idx = blockIdx.x * 256 + threadIdx.x;   // 4*64*HWPIX
  const int hw = idx & 16383;
  const int c  = (idx >> 14) & 63;
  const int n  = idx >> 20;
  const int h = hw >> 7, w = hw & 127;
  const float* xc = x  + ((size_t)n * 64 + c) * HWPIX;
  const float* f  = fw + ((size_t)n * 64 + c) * 9;
  float acc = 0.f;
#pragma unroll
  for (int i = 0; i < 3; ++i) {
    const int hh = refl128(h + i - 1);
#pragma unroll
    for (int j = 0; j < 3; ++j) {
      const int wx = refl128(w + j - 1);
      acc += f[i * 3 + j] * xc[(hh << 7) + wx];
    }
  }
  low [((size_t)n * 64 + c) * HWPIX + hw] = acc;
  high[((size_t)n * 64 + c) * HWPIX + hw] = xc[hw] - acc;
}

// pa1: 16->1 3x3 zero-pad conv + sigmoid
__global__ __launch_bounds__(256) void pa1_kernel(
    const float* __restrict__ t, const float* __restrict__ w,
    const float* __restrict__ b, float* __restrict__ out)
{
  const int idx = blockIdx.x * 256 + threadIdx.x;  // N*HWPIX
  const int hw = idx & 16383, n = idx >> 14;
  const int h = hw >> 7, wp = hw & 127;
  float acc = b[0];
  for (int ci = 0; ci < 16; ++ci) {
    const float* tc = t + ((size_t)n * 16 + ci) * HWPIX;
    for (int ky = 0; ky < 3; ++ky) {
      const int hh = h + ky - 1; if ((unsigned)hh >= 128u) continue;
      for (int kx = 0; kx < 3; ++kx) {
        const int wx = wp + kx - 1; if ((unsigned)wx >= 128u) continue;
        acc += w[ci * 9 + ky * 3 + kx] * tc[(hh << 7) + wx];
      }
    }
  }
  out[idx] = 1.f / (1.f + expf(-acc));
}

// wg / ca channel gates from gap vector (tiny 64x64 matmuls)
__global__ void gates_kernel(
    const float* __restrict__ g2,
    const float* __restrict__ wg_w, const float* __restrict__ wg_b,
    const float* __restrict__ ca0_w, const float* __restrict__ ca0_b,
    const float* __restrict__ ca1_w, const float* __restrict__ ca1_b,
    float* __restrict__ wg_out, float* __restrict__ ca_out)
{
  const int n = blockIdx.x, c = threadIdx.x;   // 64 threads
  __shared__ float t[64];
  const float* g = g2 + n * 64;
  float a0 = wg_b[c], a1 = ca0_b[c];
  for (int ci = 0; ci < 64; ++ci) { a0 += wg_w[c * 64 + ci] * g[ci];
                                    a1 += ca0_w[c * 64 + ci] * g[ci]; }
  wg_out[n * 64 + c] = 1.f / (1.f + expf(-a0));
  t[c] = geluf(a1); __syncthreads();
  float a2 = ca1_b[c];
  for (int k = 0; k < 64; ++k) a2 += ca1_w[c * 64 + k] * t[k];
  ca_out[n * 64 + c] = 1.f / (1.f + expf(-a2));
}

// cat2 strips: ca*xn2 -> [64:128], pa*xn2 -> [128:192] (pixel-major bf16)
__global__ __launch_bounds__(256) void scale_cat2_kernel(
    const float* __restrict__ xn2, const float* __restrict__ ca,
    const float* __restrict__ pa, bf16* __restrict__ outT)
{
  const int idx = blockIdx.x * 256 + threadIdx.x;
  const int hw = idx & 16383;
  const int c  = (idx >> 14) & 63;
  const int n  = idx >> 20;
  const float v = xn2[((size_t)n * 64 + c) * HWPIX + hw];
  const size_t row = ((size_t)n * HWPIX + hw) * 192;
  outT[row + 64  + c] = (bf16)(ca[n * 64 + c] * v);
  outT[row + 128 + c] = (bf16)(pa[n * HWPIX + hw] * v);
}

// fp32 OIHW weights -> bf16 [Cout][tap*Cin + ci] (tap-major K)
__global__ __launch_bounds__(256) void pack_w_kernel(
    const float* __restrict__ src, bf16* __restrict__ dst,
    int Cout, int Cin, int ksz)
{
  const int i = blockIdx.x * 256 + threadIdx.x;
  const int taps = ksz * ksz;
  if (i >= Cout * Cin * taps) return;
  const int co  = i / (Cin * taps);
  const int r   = i - co * (Cin * taps);
  const int tap = r / Cin;
  const int ci  = r - tap * Cin;
  dst[i] = (bf16)src[(co * Cin + ci) * taps + tap];
}

// ---------------------------------------------------------------------------
// Host launcher
// ---------------------------------------------------------------------------
enum {
  IN_X = 0,
  IN_FILTER_W, IN_FILTER_GN_G, IN_FILTER_GN_B,
  IN_LB0_W, IN_LB0_B, IN_LB1_A, IN_LB2_W, IN_LB2_B,
  IN_LB3_W, IN_LB3_B, IN_LB4_A, IN_LB5_W, IN_LB5_B,
  IN_LB6_W, IN_LB6_B, IN_LB7_A, IN_LB8_W, IN_LB8_B,
  IN_LB9_W, IN_LB9_B, IN_LB10_A, IN_LB11_W, IN_LB11_B,
  IN_HB_GN_G, IN_HB_GN_B, IN_HB_CONV_W, IN_HB_CONV_B,
  IN_FF0_W, IN_FF0_B, IN_FF_GN_G, IN_FF_GN_B, IN_FF3_W, IN_FF3_B,
  IN_NORM1_G, IN_NORM1_B, IN_NORM2_G, IN_NORM2_B, IN_NORM3_G, IN_NORM3_B,
  IN_CONV1_W, IN_CONV1_B, IN_CONV2_W, IN_CONV2_B,
  IN_ES_W, IN_ES_B, IN_EM_W, IN_EM_B, IN_EL_W, IN_EL_B,
  IN_WV0_W, IN_WV0_B, IN_WV1_W, IN_WV1_B, IN_WG_W, IN_WG_B,
  IN_CA0_W, IN_CA0_B, IN_CA1_W, IN_CA1_B,
  IN_PA0_W, IN_PA0_B, IN_PA1_W, IN_PA1_B,
  IN_MLP0_W, IN_MLP0_B, IN_MLP_GN_G, IN_MLP_GN_B, IN_MLP3_W, IN_MLP3_B,
  IN_MLP2_0_W, IN_MLP2_0_B, IN_MLP2_GN_G, IN_MLP2_GN_B, IN_MLP2_3_W, IN_MLP2_3_B,
  IN_DF0_W, IN_DF0_B, IN_DF_GN_G, IN_DF_GN_B, IN_DF3_W, IN_DF3_B
};

extern "C" void kernel_launch(void* const* d_in, const int* in_sizes, int n_in,
                              void* d_out, int out_size, void* d_ws, size_t ws_size,
                              hipStream_t stream) {
  (void)in_sizes; (void)n_in; (void)out_size; (void)ws_size;
  char* ws = (char*)d_ws;
  const size_t MB = 1ull << 20;

  // fp32 arena
  float* FAp = (float*)(ws + 0 * MB);      // xn2 / xn3
  float* FBp = (float*)(ws + 16 * MB);     // xpr / wv
  float* FCp = (float*)(ws + 32 * MB);     // base
  float* FDp = (float*)(ws + 48 * MB);     // low
  float* FEp = (float*)(ws + 64 * MB);     // high
  float* FGp = (float*)(ws + 80 * MB);     // x2
  float* FHp = (float*)(ws + 96 * MB);     // x3
  float* FIp = (float*)(ws + 112 * MB);    // 128-ch fp32 (u/m)
  float* FJp = (float*)(ws + 144 * MB);    // 256-ch fp32 (m2)
  float* FKp = (float*)(ws + 208 * MB);    // 16-ch scratch a
  float* FLp = (float*)(ws + 212 * MB);    // 16-ch scratch b
  char*  G   =          ws + 216 * MB;     // small vectors + bf16 weights
  // bf16 pixel-major arena
  bf16* X0p = (bf16*)(ws + 224 * MB);      // 64-ch XT a
  bf16* X1p = (bf16*)(ws + 232 * MB);      // 64-ch XT b (xpr / hb)
  bf16* X2p = (bf16*)(ws + 240 * MB);      // 320-ch cat1
  bf16* X3p = (bf16*)(ws + 280 * MB);      // 192-ch cat2 / 128-ch dfc
  bf16* X4p = (bf16*)(ws + 304 * MB);      // 128-ch fused
  bf16* X5p = (bf16*)(ws + 320 * MB);      // 128-ch XT (post-GN)
  bf16* X6p = (bf16*)(ws + 336 * MB);      // 256-ch XT

  // small-vector sub-arena
  float* GAPV = (float*)(G + 0x0000);      // [N*64]
  float* FW9  = (float*)(G + 0x1000);      // [N*64*9]
  float* WGv  = (float*)(G + 0x4000);      // [N*64]
  float* CAv  = (float*)(G + 0x5000);      // [N*64]
  float* PAv  = (float*)(G + 0x8000);      // [N*HW]
  bf16* WBconv1 = (bf16*)(G + 0x50000);
  bf16* WBwv0   = (bf16*)(G + 0x52000);
  bf16* WBff0   = (bf16*)(G + 0x54000);
  bf16* WBff3   = (bf16*)(G + 0x5C000);
  bf16* WBmlp3  = (bf16*)(G + 0x60000);
  bf16* WBdf0   = (bf16*)(G + 0x64000);
  bf16* WBdf3   = (bf16*)(G + 0x6C000);
  bf16* WBmlp0  = (bf16*)(G + 0x70000);
  bf16* WBm20   = (bf16*)(G + 0x84000);
  bf16* WBm23   = (bf16*)(G + 0x9C000);
  bf16* WBhb    = (bf16*)(G + 0xA4000);
  bf16* WBconv2 = (bf16*)(G + 0xB6000);

#define F(i) ((const float*)d_in[i])

  auto pack = [&](int idx, bf16* dst, int Cout, int Cin, int ksz) {
    const int total = Cout * Cin * ksz * ksz;
    pack_w_kernel<<<dim3((total + 255) / 256), dim3(256), 0, stream>>>(
        F(idx), dst, Cout, Cin, ksz);
  };
  auto gemm = [&](const bf16* A, const bf16* Bt, int ldb, int boff, int K, int Cout,
                  const float* bias, const float* resid,
                  float* outF, int outFCtot, bf16* outT, int ldt, int toff) {
    gemm1x1_wmma_kernel<<<dim3(QTOT / 256, Cout / 16), dim3(128), 0, stream>>>(
        A, Bt, ldb, boff, K, bias, resid, outF, outFCtot, outT, ldt, toff);
  };
  auto gn = [&](const float* in, int C, int cpg, const float* g_, const float* b_,
                int act, float* outF, bf16* outT, int ldt, int toff) {
    groupnorm_kernel<<<dim3(NB * (C / cpg)), dim3(256), 0, stream>>>(
        in, C, cpg, g_, b_, act, outF, outT, ldt, toff);
  };
  auto dw = [&](const float* in, int C, const float* w, const float* b,
                int ksz, int dil, int pad, int refl,
                const float* resid, const float* scale,
                float* outF, bf16* outT, int ldt, int toff) {
    const int total = NB * C * HWPIX;
    dwconv_kernel<<<dim3((total + 255) / 256), dim3(256), 0, stream>>>(
        in, C, w, b, ksz, dil, pad, refl, resid, scale, outF, outT, ldt, toff);
  };
  auto c1s = [&](const float* in, int Cin, int Cout, const float* w, const float* b,
                 int act, const float* pa, float* outF, bf16* outT, int ldt, int toff) {
    const int total = NB * Cout * HWPIX;
    conv1x1_small_kernel<<<dim3((total + 255) / 256), dim3(256), 0, stream>>>(
        in, Cin, Cout, w, b, act, pa, outF, outT, ldt, toff);
  };

  // frequency-processing block (shared params; called in stage 1 and stage 3)
  auto freq_block = [&](const float* xin, bf16* outT, int out_ldt, int out_toff) {
    gap_kernel<<<dim3(NB * 64), dim3(256), 0, stream>>>(xin, GAPV);
    freq_filter_kernel<<<dim3(NB), dim3(576), 0, stream>>>(
        GAPV, F(IN_FILTER_W), F(IN_FILTER_GN_G), F(IN_FILTER_GN_B), FW9);
    lowhigh_kernel<<<dim3(NB * 64 * HWPIX / 256), dim3(256), 0, stream>>>(
        xin, FW9, FDp, FEp);
    // low branch: 16-ch bottleneck chain (bandwidth-bound VALU)
    c1s(FDp, 64, 16, F(IN_LB0_W), F(IN_LB0_B), 2, F(IN_LB1_A), FKp, nullptr, 0, 0);
    dw(FKp, 16, F(IN_LB2_W), F(IN_LB2_B), 3, 5, 5, 0, nullptr, nullptr, FLp, nullptr, 0, 0);
    c1s(FLp, 16, 16, F(IN_LB3_W), F(IN_LB3_B), 2, F(IN_LB4_A), FKp, nullptr, 0, 0);
    dw(FKp, 16, F(IN_LB5_W), F(IN_LB5_B), 3, 1, 1, 0, nullptr, nullptr, FLp, nullptr, 0, 0);
    c1s(FLp, 16, 16, F(IN_LB6_W), F(IN_LB6_B), 2, F(IN_LB7_A), FKp, nullptr, 0, 0);
    dw(FKp, 16, F(IN_LB8_W), F(IN_LB8_B), 3, 3, 3, 0, nullptr, nullptr, FLp, nullptr, 0, 0);
    c1s(FLp, 16, 16, F(IN_LB9_W), F(IN_LB9_B), 2, F(IN_LB10_A), FKp, nullptr, 0, 0);
    c1s(FKp, 16, 64, F(IN_LB11_W), F(IN_LB11_B), 0, nullptr,
        nullptr, X4p, 128, 0);                               // enhanced_low -> fused[:,0:64]
    // high branch: GN(4)+leaky -> 3x3 dense WMMA conv (+high residual)
    gn(FEp, 64, 16, F(IN_HB_GN_G), F(IN_HB_GN_B), 2, nullptr, X1p, 64, 0);
    gemm_im2col_wmma_kernel<3, 2><<<dim3(QTOT / 256, 4), dim3(128), 0, stream>>>(
        WBhb, X1p, 64, 0, 576, 64, 1, 1, F(IN_HB_CONV_B), FEp,
        nullptr, 0, X4p, 128, 64);                           // enhanced_high -> fused[:,64:128]
    // fusion FFN: ff0 -> GN(4)+gelu -> ff3
    gemm(WBff0, X4p, 128, 0, 128, 128, F(IN_FF0_B), nullptr,
         FIp, 128, nullptr, 0, 0);
    gn(FIp, 128, 32, F(IN_FF_GN_G), F(IN_FF_GN_B), 1, nullptr, X5p, 128, 0);
    gemm(WBff3, X5p, 128, 0, 128, 64, F(IN_FF3_B), nullptr,
         nullptr, 0, outT, out_ldt, out_toff);
  };

  // ---- pack all dense-conv weights to bf16 (tap-major) ----
  pack(IN_CONV1_W, WBconv1, 64, 64, 1);
  pack(IN_CONV2_W, WBconv2, 64, 64, 5);
  pack(IN_HB_CONV_W, WBhb, 64, 64, 3);
  pack(IN_FF0_W, WBff0, 128, 128, 1);
  pack(IN_FF3_W, WBff3, 64, 128, 1);
  pack(IN_MLP0_W, WBmlp0, 128, 320, 1);
  pack(IN_MLP3_W, WBmlp3, 64, 128, 1);
  pack(IN_WV0_W, WBwv0, 64, 64, 1);
  pack(IN_MLP2_0_W, WBm20, 256, 192, 1);
  pack(IN_MLP2_3_W, WBm23, 64, 256, 1);
  pack(IN_DF0_W, WBdf0, 128, 128, 1);
  pack(IN_DF3_W, WBdf3, 64, 128, 1);

  const float* x = F(IN_X);

  // ---- stage 1: multi-scale spectral mixing ----
  gn(x, 64, 2, F(IN_NORM1_G), F(IN_NORM1_B), 0, nullptr, X0p, 64, 0);
  gemm(WBconv1, X0p, 64, 0, 64, 64, F(IN_CONV1_B), nullptr,
       FBp, 64, X1p, 64, 0);                                  // xpr
  gemm_im2col_wmma_kernel<5, 1><<<dim3(QTOT / 256, 4), dim3(128), 0, stream>>>(
      WBconv2, X1p, 64, 0, 1600, 64, 1, 2, F(IN_CONV2_B), nullptr,
      FCp, 64, X2p, 320, 0);                                  // base -> cat1[:,0:64]
  freq_block(FBp, X2p, 320, 256);                             // freq -> cat1[:,256:320]
  dw(FCp, 64, F(IN_ES_W), F(IN_ES_B), 3, 3, 3, 1, FCp, nullptr, nullptr, X2p, 320, 64);
  dw(FCp, 64, F(IN_EM_W), F(IN_EM_B), 5, 3, 6, 1, FCp, nullptr, nullptr, X2p, 320, 128);
  dw(FCp, 64, F(IN_EL_W), F(IN_EL_B), 7, 3, 9, 1, FCp, nullptr, nullptr, X2p, 320, 192);
  gemm(WBmlp0, X2p, 320, 0, 320, 128, F(IN_MLP0_B), nullptr,
       FIp, 128, nullptr, 0, 0);
  gn(FIp, 128, 4, F(IN_MLP_GN_G), F(IN_MLP_GN_B), 1, nullptr, X5p, 128, 0);
  gemm(WBmlp3, X5p, 128, 0, 128, 64, F(IN_MLP3_B), x,
       FGp, 64, nullptr, 0, 0);                               // x2

  // ---- stage 2: attention gating ----
  gn(FGp, 64, 2, F(IN_NORM2_G), F(IN_NORM2_B), 0, FAp, X0p, 64, 0);
  gemm(WBwv0, X0p, 64, 0, 64, 64, F(IN_WV0_B), nullptr,
       FBp, 64, nullptr, 0, 0);                               // wv pre-depthwise
  gap_kernel<<<dim3(NB * 64), dim3(256), 0, stream>>>(FAp, GAPV);
  gates_kernel<<<dim3(NB), dim3(64), 0, stream>>>(
      GAPV, F(IN_WG_W), F(IN_WG_B), F(IN_CA0_W), F(IN_CA0_B),
      F(IN_CA1_W), F(IN_CA1_B), WGv, CAv);
  dw(FBp, 64, F(IN_WV1_W), F(IN_WV1_B), 3, 1, 1, 1, nullptr, WGv,
     nullptr, X3p, 192, 0);                                   // wv*wg -> cat2[:,0:64]
  c1s(FAp, 64, 16, F(IN_PA0_W), F(IN_PA0_B), 1, nullptr, FKp, nullptr, 0, 0);
  pa1_kernel<<<dim3(NB * HWPIX / 256), dim3(256), 0, stream>>>(
      FKp, F(IN_PA1_W), F(IN_PA1_B), PAv);
  scale_cat2_kernel<<<dim3(NB * 64 * HWPIX / 256), dim3(256), 0, stream>>>(
      FAp, CAv, PAv, X3p);                                    // cat2[:,64:192]
  gemm(WBm20, X3p, 192, 0, 192, 256, F(IN_MLP2_0_B), nullptr,
       FJp, 256, nullptr, 0, 0);
  gn(FJp, 256, 8, F(IN_MLP2_GN_G), F(IN_MLP2_GN_B), 1, nullptr, X6p, 256, 0);
  gemm(WBm23, X6p, 256, 0, 256, 64, F(IN_MLP2_3_B), FGp,
       FHp, 64, nullptr, 0, 0);                               // x3

  // ---- stage 3: domain fusion ----
  gn(FHp, 64, 2, F(IN_NORM3_G), F(IN_NORM3_B), 0, FAp, X3p, 128, 0);  // dfc[:,0:64]
  freq_block(FAp, X3p, 128, 64);                              // freq2 -> dfc[:,64:128]
  gemm(WBdf0, X3p, 128, 0, 128, 128, F(IN_DF0_B), nullptr,
       FIp, 128, nullptr, 0, 0);
  gn(FIp, 128, 4, F(IN_DF_GN_G), F(IN_DF_GN_B), 1, nullptr, X5p, 128, 0);
  gemm(WBdf3, X5p, 128, 0, 128, 64, F(IN_DF3_B), FHp,
       (float*)d_out, 64, nullptr, 0, 0);                     // out = x3 + df3(u)
#undef F
}
